// TransactionLongformer_79611513799350
// MI455X (gfx1250) — matrix operand order repaired and checked by reading, not compiled, not run on hardware
//
#include <hip/hip_runtime.h>
#include <hip/hip_bf16.h>
#include <stdint.h>

namespace {

constexpr int kB = 2, kS = 4096, kF = 128, kHD = 768, kL = 4, kNH = 8,
              kDH = 96, kW1 = 256, kFF = 3072, kOUT = 512;
constexpr int kTok = kB * kS;      // 8192 tokens
constexpr int kQKV = 3 * kHD;      // 2304
constexpr int kGKV = 2 * kHD;      // 1536 (gate k,v)
constexpr float kScale = 0.10206207261596577f;  // 1/sqrt(96)

typedef __attribute__((ext_vector_type(16))) __bf16 bf16x16;
typedef __attribute__((ext_vector_type(8)))  float  f32x8;

union Frag16 { bf16x16 v; uint4 q[2]; };

__device__ __forceinline__ uint16_t f2bf(float f) {
  uint32_t u = __float_as_uint(f);
  uint32_t r = u + 0x7FFFu + ((u >> 16) & 1u);   // round-to-nearest-even
  return (uint16_t)(r >> 16);
}
__device__ __forceinline__ float bf2f(uint16_t h) {
  return __uint_as_float(((uint32_t)h) << 16);
}

__device__ __forceinline__ f32x8 wmma_bf16(const Frag16& a, const Frag16& b, f32x8 c) {
  return __builtin_amdgcn_wmma_f32_16x16x32_bf16(false, a.v, false, b.v,
                                                 (short)0, c, false, false);
}

// CDNA5 async global->LDS copy (ASYNCcnt tracked), 16B per lane.
__device__ __forceinline__ void async_b128(uint32_t lds_off, const void* gptr) {
  asm volatile("global_load_async_to_lds_b128 %0, %1, off"
               :: "v"(lds_off), "v"(gptr) : "memory");
}
__device__ __forceinline__ void wait_async0() {
  asm volatile("s_wait_asynccnt 0" ::: "memory");
}

// A-fragment (16x32 bf16, MxK): lane L holds row L&15, K range (L>>4)*16 .. +15
// contiguous.
__device__ __forceinline__ void ld_afrag(Frag16& f, const uint16_t* p0, int ld) {
  const int lane = threadIdx.x & 31;
  const uint16_t* p = p0 + (size_t)(lane & 15) * ld + ((lane >> 4) << 4);
  f.q[0] = *(const uint4*)(p);
  f.q[1] = *(const uint4*)(p + 8);
}
// B-fragment (32x16 bf16, KxN) sourced from B^T (N rows x K cols): lane L holds
// column N=L&15, K range (L>>4)*16 .. +15 contiguous.
__device__ __forceinline__ void ld_bfrag(Frag16& f, const uint16_t* p0, int ld) {
  const int lane = threadIdx.x & 31;
  const uint16_t* p = p0 + (size_t)(lane & 15) * ld + ((lane >> 4) << 4);
  f.q[0] = *(const uint4*)(p);
  f.q[1] = *(const uint4*)(p + 8);
}

// ---------------------------------------------------------------------------
// Generic bf16 WMMA GEMM: C[M,N] = A[M,K] @ B[K,N] + bias, B given transposed.
// OMODE: 0 = f32 out, 1 = bf16 out, 2 = bf16 out + exact GELU.
// Block 256 threads (8 wave32), block tile 64x256, wave tile 32x64, K step 32.
// Double-buffered async global->LDS staging.
// ---------------------------------------------------------------------------
template<int OMODE>
__global__ __launch_bounds__(256) void gemm_bf16(
    const uint16_t* __restrict__ A, const uint16_t* __restrict__ Bt,
    const float* __restrict__ bias, void* __restrict__ Cout,
    int M, int N, int K) {
  (void)M;
  __shared__ __align__(16) uint16_t sA[2][64 * 32];
  __shared__ __align__(16) uint16_t sB[2][256 * 32];
  const int tid = threadIdx.x;
  const int wid = tid >> 5, lane = tid & 31;
  const int m0 = blockIdx.y * 64, n0 = blockIdx.x * 256;
  const int wm = (wid & 1) * 32, wn = (wid >> 1) * 64;
  const uint32_t sAoff = (uint32_t)(uintptr_t)(&sA[0][0]);
  const uint32_t sBoff = (uint32_t)(uintptr_t)(&sB[0][0]);

  auto issue = [&](int buf, int k0) {
    // A tile: 64 rows x 32 cols bf16 = 4KB, one 16B async load per thread.
    int row = tid >> 2, seg = tid & 3;
    async_b128(sAoff + (uint32_t)buf * (64 * 32 * 2) + (uint32_t)(row * 64 + seg * 16),
               A + (size_t)(m0 + row) * K + k0 + seg * 8);
    // B tile: 256 rows x 32 cols bf16 = 16KB, four 16B async loads per thread.
    const uint16_t* g = Bt + (size_t)(n0 + tid) * K + k0;
    uint32_t l = sBoff + (uint32_t)buf * (256 * 32 * 2) + (uint32_t)tid * 64;
#pragma unroll
    for (int s2 = 0; s2 < 4; ++s2) async_b128(l + s2 * 16, g + s2 * 8);
  };

  const f32x8 z8 = {0.f, 0.f, 0.f, 0.f, 0.f, 0.f, 0.f, 0.f};
  f32x8 acc[2][4];
#pragma unroll
  for (int i = 0; i < 2; ++i)
#pragma unroll
    for (int j = 0; j < 4; ++j) acc[i][j] = z8;

  issue(0, 0);
  int buf = 0;
  for (int k0 = 0; k0 < K; k0 += 32) {
    wait_async0();
    __syncthreads();
    if (k0 + 32 < K) issue(buf ^ 1, k0 + 32);
    Frag16 af[2], bfv[4];
#pragma unroll
    for (int mt = 0; mt < 2; ++mt) ld_afrag(af[mt], &sA[buf][(wm + mt * 16) * 32], 32);
#pragma unroll
    for (int nt = 0; nt < 4; ++nt) ld_bfrag(bfv[nt], &sB[buf][(wn + nt * 16) * 32], 32);
#pragma unroll
    for (int mt = 0; mt < 2; ++mt)
#pragma unroll
      for (int nt = 0; nt < 4; ++nt)
        acc[mt][nt] = wmma_bf16(af[mt], bfv[nt], acc[mt][nt]);
    buf ^= 1;
  }

  const int hi = lane >> 4, nl = lane & 15;
#pragma unroll
  for (int mt = 0; mt < 2; ++mt)
#pragma unroll
    for (int nt = 0; nt < 4; ++nt)
#pragma unroll
      for (int r = 0; r < 8; ++r) {
        int m = m0 + wm + mt * 16 + r + hi * 8;
        int n = n0 + wn + nt * 16 + nl;
        float v = acc[mt][nt][r] + bias[n];
        if (OMODE == 2) v = 0.5f * v * (1.0f + erff(v * 0.70710678118654752f));
        if (OMODE == 0) ((float*)Cout)[(size_t)m * N + n] = v;
        else            ((uint16_t*)Cout)[(size_t)m * N + n] = f2bf(v);
      }
}

// ---------------------------------------------------------------------------
// Banded (sliding-chunk) attention, flash-style single pass. Block = 128 thr
// (4 waves); wave handles 32 queries of one chunk; loops 32-key tiles over the
// 3*W1 window. Scores and P@V both via bf16 WMMA; online softmax with 16-lane
// shuffle reductions; global-token column folded in at the end.
// grid: (S/128, NH, B)
// ---------------------------------------------------------------------------
__global__ __launch_bounds__(128) void attn_local(
    const uint16_t* __restrict__ qkv, uint16_t* __restrict__ aout) {
  __shared__ __align__(16) uint16_t Vt[kDH * 32];      // V tile transposed (feat x key)
  __shared__ __align__(16) uint16_t Pbuf[4][32 * 32];  // per-wave P relayout
  const int tid = threadIdx.x, wid = tid >> 5, lane = tid & 31;
  const int hh = blockIdx.y, b = blockIdx.z;
  const int q0 = blockIdx.x * 128;   // block query base (chunk aligned, 128|256)
  const int c = q0 / kW1;
  const int wq = q0 + wid * 32;      // wave query base (global)
  const int iq = wq - c * kW1;       // chunk-local base
  const int kbase = c * kW1 - kW1;   // keypos of window index j=0
  const int hi = lane >> 4, nl = lane & 15;

  const uint16_t* Qb = qkv + ((size_t)(b * kS + wq)) * kQKV + hh * kDH;
  const uint16_t* Kb = qkv + ((size_t)(b * kS)) * kQKV + kHD + hh * kDH;
  const uint16_t* Vb = qkv + ((size_t)(b * kS)) * kQKV + 2 * kHD + hh * kDH;

  Frag16 aq[2][3];
#pragma unroll
  for (int mt = 0; mt < 2; ++mt)
#pragma unroll
    for (int kk = 0; kk < 3; ++kk)
      ld_afrag(aq[mt][kk], Qb + (size_t)(mt * 16) * kQKV + kk * 32, kQKV);

  const f32x8 z8 = {0.f, 0.f, 0.f, 0.f, 0.f, 0.f, 0.f, 0.f};
  f32x8 o[2][6];
  float mrow[2][8], lrow[2][8];
#pragma unroll
  for (int mt = 0; mt < 2; ++mt) {
#pragma unroll
    for (int nt = 0; nt < 6; ++nt) o[mt][nt] = z8;
#pragma unroll
    for (int r = 0; r < 8; ++r) { mrow[mt][r] = -1.0e30f; lrow[mt][r] = 0.f; }
  }

  for (int j0 = 0; j0 < 3 * kW1; j0 += 32) {
    __syncthreads();
    // Stage V tile transposed into LDS (cooperative, coalesced global reads).
    for (int e = tid; e < kDH * 32; e += 128) {
      int feat = e % kDH, key = e / kDH;
      int kp = kbase + j0 + key;
      kp = kp < 0 ? 0 : (kp > kS - 1 ? kS - 1 : kp);
      Vt[feat * 32 + key] = Vb[(size_t)kp * kQKV + feat];
    }
    __syncthreads();

    // Scores: S(32q x 32k) = Q(32x96) @ K^T(96x32)
    f32x8 sc[2][2];
    sc[0][0] = z8; sc[0][1] = z8; sc[1][0] = z8; sc[1][1] = z8;
#pragma unroll
    for (int kk = 0; kk < 3; ++kk) {
      Frag16 bk[2];
#pragma unroll
      for (int ntk = 0; ntk < 2; ++ntk) {
        int kp = kbase + j0 + ntk * 16 + nl;
        kp = kp < 0 ? 0 : (kp > kS - 1 ? kS - 1 : kp);
        const uint16_t* p = Kb + (size_t)kp * kQKV + kk * 32 + hi * 16;
        bk[ntk].q[0] = *(const uint4*)(p);
        bk[ntk].q[1] = *(const uint4*)(p + 8);
      }
#pragma unroll
      for (int mt = 0; mt < 2; ++mt)
#pragma unroll
        for (int ntk = 0; ntk < 2; ++ntk)
          sc[mt][ntk] = wmma_bf16(aq[mt][kk], bk[ntk], sc[mt][ntk]);
    }

    // Mask + online softmax + relayout P into LDS.
#pragma unroll
    for (int mt = 0; mt < 2; ++mt) {
#pragma unroll
      for (int r = 0; r < 8; ++r) {
        int irow = iq + mt * 16 + r + hi * 8;   // chunk-local query index
        float vv[2];
#pragma unroll
        for (int ntk = 0; ntk < 2; ++ntk) {
          int j = j0 + ntk * 16 + nl;
          int rel = j - irow;
          int kp = kbase + j;
          bool ok = (rel >= 0) && (rel <= 2 * kW1) && (kp >= 0) && (kp < kS) && (kp != 0);
          vv[ntk] = ok ? sc[mt][ntk][r] : -1.0e9f;
        }
        float tmax = fmaxf(vv[0], vv[1]);
        tmax = fmaxf(tmax, __shfl_xor(tmax, 1, 32));
        tmax = fmaxf(tmax, __shfl_xor(tmax, 2, 32));
        tmax = fmaxf(tmax, __shfl_xor(tmax, 4, 32));
        tmax = fmaxf(tmax, __shfl_xor(tmax, 8, 32));
        float mnew = fmaxf(mrow[mt][r], tmax);
        float corr = __expf(mrow[mt][r] - mnew);
        mrow[mt][r] = mnew;
        float p0 = __expf(vv[0] - mnew);
        float p1 = __expf(vv[1] - mnew);
        float rs = p0 + p1;
        rs += __shfl_xor(rs, 1, 32);
        rs += __shfl_xor(rs, 2, 32);
        rs += __shfl_xor(rs, 4, 32);
        rs += __shfl_xor(rs, 8, 32);
        lrow[mt][r] = lrow[mt][r] * corr + rs;
#pragma unroll
        for (int nt = 0; nt < 6; ++nt) o[mt][nt][r] *= corr;
        int prow = mt * 16 + r + hi * 8;
        Pbuf[wid][prow * 32 + nl]      = f2bf(p0);
        Pbuf[wid][prow * 32 + 16 + nl] = f2bf(p1);
      }
    }

    // O += P(32x32) @ V(32x96)
    Frag16 ap[2], bv[6];
#pragma unroll
    for (int mt = 0; mt < 2; ++mt) ld_afrag(ap[mt], &Pbuf[wid][mt * 16 * 32], 32);
#pragma unroll
    for (int nt = 0; nt < 6; ++nt) {
      const uint16_t* p = &Vt[(nt * 16 + nl) * 32 + hi * 16];
      bv[nt].q[0] = *(const uint4*)(p);
      bv[nt].q[1] = *(const uint4*)(p + 8);
    }
#pragma unroll
    for (int mt = 0; mt < 2; ++mt)
#pragma unroll
      for (int nt = 0; nt < 6; ++nt)
        o[mt][nt] = wmma_bf16(ap[mt], bv[nt], o[mt][nt]);
  }

  // Global-token column: score vs K[token 0], prob * V[token 0].
#pragma unroll
  for (int mt = 0; mt < 2; ++mt)
#pragma unroll
    for (int r = 0; r < 8; ++r) {
      int lr = mt * 16 + r + hi * 8;
      const uint16_t* qr = Qb + (size_t)lr * kQKV;
      float g = 0.f;
      for (int d = 0; d < kDH; ++d) g += bf2f(qr[d]) * bf2f(Kb[d]);
      float mnew = fmaxf(mrow[mt][r], g);
      float corr = __expf(mrow[mt][r] - mnew);
      float pg = __expf(g - mnew);
      lrow[mt][r] = lrow[mt][r] * corr + pg;
#pragma unroll
      for (int nt = 0; nt < 6; ++nt) {
        float v0 = bf2f(Vb[nt * 16 + nl]);
        o[mt][nt][r] = o[mt][nt][r] * corr + pg * v0;
      }
    }

  // Normalize and store (row 0 later overwritten by global-attention kernel).
#pragma unroll
  for (int mt = 0; mt < 2; ++mt)
#pragma unroll
    for (int nt = 0; nt < 6; ++nt)
#pragma unroll
      for (int r = 0; r < 8; ++r) {
        int qpos = wq + mt * 16 + r + hi * 8;
        float val = o[mt][nt][r] / lrow[mt][r];
        aout[((size_t)(b * kS + qpos)) * kHD + hh * kDH + nt * 16 + nl] = f2bf(val);
      }
}

// Global attention for token 0 (one block per (head, batch)).
__global__ __launch_bounds__(256) void attn_global(
    const float* __restrict__ qg, const uint16_t* __restrict__ gkv,
    uint16_t* __restrict__ aout) {
  __shared__ float sc[kS];
  __shared__ float red[256];
  const int tid = threadIdx.x, hh = blockIdx.x, b = blockIdx.y;
  const float* q = qg + b * kHD + hh * kDH;
  const uint16_t* kgp = gkv + (size_t)(b * kS) * kGKV + hh * kDH;
  const uint16_t* vgp = kgp + kHD;
  float lmax = -1.0e30f;
  for (int j = tid; j < kS; j += 256) {
    const uint16_t* kr = kgp + (size_t)j * kGKV;
    float s = 0.f;
    for (int d = 0; d < kDH; ++d) s += q[d] * bf2f(kr[d]);
    sc[j] = s;
    lmax = fmaxf(lmax, s);
  }
  red[tid] = lmax; __syncthreads();
  for (int o2 = 128; o2; o2 >>= 1) { if (tid < o2) red[tid] = fmaxf(red[tid], red[tid + o2]); __syncthreads(); }
  float m = red[0]; __syncthreads();
  float ls = 0.f;
  for (int j = tid; j < kS; j += 256) ls += __expf(sc[j] - m);
  red[tid] = ls; __syncthreads();
  for (int o2 = 128; o2; o2 >>= 1) { if (tid < o2) red[tid] += red[tid + o2]; __syncthreads(); }
  float Z = red[0]; __syncthreads();
  if (tid < kDH) {
    float o = 0.f;
    for (int j = 0; j < kS; ++j) o += __expf(sc[j] - m) * bf2f(vgp[(size_t)j * kGKV + tid]);
    aout[(size_t)(b * kS) * kHD + hh * kDH + tid] = f2bf(o / Z);
  }
}

// Embedding (K=128, 0.3% of FLOPs -> plain VALU).
__global__ __launch_bounds__(256) void embed_kernel(
    const float* __restrict__ x, const float* __restrict__ emb_w,
    const float* __restrict__ emb_b, const float* __restrict__ pos_emb,
    const float* __restrict__ hf_pos, const float* __restrict__ tok_emb,
    float* __restrict__ out) {
  const int t = blockIdx.x, s = t % kS;
  __shared__ float xr[kF];
  for (int i = threadIdx.x; i < kF; i += 256) xr[i] = x[(size_t)t * kF + i];
  __syncthreads();
  for (int n = threadIdx.x; n < kHD; n += 256) {
    float a = 0.f;
    for (int k = 0; k < kF; ++k) a += xr[k] * emb_w[(size_t)k * kHD + n];
    out[(size_t)t * kHD + n] = a + emb_b[n] + pos_emb[(size_t)s * kHD + n] +
                               hf_pos[(size_t)s * kHD + n] + tok_emb[n];
  }
}

// LayerNorm over last dim (768); optional residual add; writes f32 + bf16.
__global__ __launch_bounds__(256) void ln_kernel(
    const float* __restrict__ x, const float* __restrict__ add,
    const float* __restrict__ g, const float* __restrict__ bb, float eps,
    float* __restrict__ outf, uint16_t* __restrict__ outb) {
  const int t = blockIdx.x, tid = threadIdx.x;
  __shared__ float row[kHD];
  __shared__ float red[256];
  const float* xr = x + (size_t)t * kHD;
  float s = 0.f;
  for (int i = tid; i < kHD; i += 256) {
    float v = xr[i];
    if (add) v += add[(size_t)t * kHD + i];
    row[i] = v; s += v;
  }
  red[tid] = s; __syncthreads();
  for (int o2 = 128; o2; o2 >>= 1) { if (tid < o2) red[tid] += red[tid + o2]; __syncthreads(); }
  float mean = red[0] / kHD; __syncthreads();
  float vs = 0.f;
  for (int i = tid; i < kHD; i += 256) { float d = row[i] - mean; vs += d * d; }
  red[tid] = vs; __syncthreads();
  for (int o2 = 128; o2; o2 >>= 1) { if (tid < o2) red[tid] += red[tid + o2]; __syncthreads(); }
  float inv = rsqrtf(red[0] / kHD + eps); __syncthreads();
  for (int i = tid; i < kHD; i += 256) {
    float o = (row[i] - mean) * inv * g[i] + bb[i];
    outf[(size_t)t * kHD + i] = o;
    outb[(size_t)t * kHD + i] = f2bf(o);
  }
}

// Gate query for token 0 (2 rows only).
__global__ void qg_kernel(const float* __restrict__ h, const float* __restrict__ Wg0,
                          const float* __restrict__ bg0, float* __restrict__ qg) {
  const int b = blockIdx.x, n = threadIdx.x;
  const float* hr = h + (size_t)b * kS * kHD;
  float a = 0.f;
  for (int k = 0; k < kHD; ++k) a += hr[k] * Wg0[(size_t)k * kHD + n];
  qg[b * kHD + n] = (a + bg0[n]) * kScale;
}

// Fused head: p1 GEMM (2x768) + LN(1e-5) + relu + p2 GEMM (2x512).
__global__ __launch_bounds__(256) void head_kernel(
    const float* __restrict__ hf, const float* __restrict__ p1w,
    const float* __restrict__ p1b, const float* __restrict__ png,
    const float* __restrict__ pnb, const float* __restrict__ p2w,
    const float* __restrict__ p2b, float* __restrict__ out) {
  const int b = blockIdx.x, tid = threadIdx.x;
  __shared__ float y[kHD];
  __shared__ float red[256];
  const float* cls = hf + (size_t)b * kS * kHD;
  for (int n = tid; n < kHD; n += 256) {
    float a = 0.f;
    for (int k = 0; k < kHD; ++k) a += cls[k] * p1w[(size_t)k * kHD + n];
    y[n] = a + p1b[n];
  }
  __syncthreads();
  float s = 0.f;
  for (int n = tid; n < kHD; n += 256) s += y[n];
  red[tid] = s; __syncthreads();
  for (int o2 = 128; o2; o2 >>= 1) { if (tid < o2) red[tid] += red[tid + o2]; __syncthreads(); }
  float mean = red[0] / kHD; __syncthreads();
  float vs = 0.f;
  for (int n = tid; n < kHD; n += 256) { float d = y[n] - mean; vs += d * d; }
  red[tid] = vs; __syncthreads();
  for (int o2 = 128; o2; o2 >>= 1) { if (tid < o2) red[tid] += red[tid + o2]; __syncthreads(); }
  float inv = rsqrtf(red[0] / kHD + 1e-5f); __syncthreads();
  for (int n = tid; n < kHD; n += 256)
    y[n] = fmaxf(0.f, (y[n] - mean) * inv * png[n] + pnb[n]);
  __syncthreads();
  for (int n = tid; n < kOUT; n += 256) {
    float a = 0.f;
    for (int k = 0; k < kHD; ++k) a += y[k] * p2w[(size_t)k * kOUT + n];
    out[(size_t)b * kOUT + n] = a + p2b[n];
  }
}

// Repack f32 [K][N] weight -> bf16 [N][K] (B-transposed) with optional scale.
__global__ void repack_kernel(const float* __restrict__ src, uint16_t* __restrict__ dst,
                              int Kd, int Nd, float scale) {
  size_t i = (size_t)blockIdx.x * 256 + threadIdx.x;
  if (i >= (size_t)Kd * Nd) return;
  int k = (int)(i / Nd), n = (int)(i % Nd);
  dst[(size_t)n * Kd + k] = f2bf(src[i] * scale);
}

// Scale the q-third of the fused qkv bias by 1/sqrt(DH).
__global__ void qkv_bias_kernel(const float* __restrict__ bqkv, float* __restrict__ dst) {
  int i = blockIdx.x * 256 + threadIdx.x;
  if (i >= kL * kQKV) return;
  int j = i % kQKV;
  dst[i] = bqkv[i] * (j < kHD ? kScale : 1.f);
}

}  // namespace

extern "C" void kernel_launch(void* const* d_in, const int* in_sizes, int n_in,
                              void* d_out, int out_size, void* d_ws, size_t ws_size,
                              hipStream_t stream) {
  (void)in_sizes; (void)n_in; (void)out_size; (void)ws_size;
  const float* x        = (const float*)d_in[0];
  const float* emb_w    = (const float*)d_in[1];
  const float* emb_b    = (const float*)d_in[2];
  const float* pos_emb  = (const float*)d_in[3];
  const float* hf_pos   = (const float*)d_in[4];
  const float* tok_emb  = (const float*)d_in[5];
  const float* eln_g    = (const float*)d_in[6];
  const float* eln_b    = (const float*)d_in[7];
  const float* Wqkv     = (const float*)d_in[8];
  const float* bqkv     = (const float*)d_in[9];
  const float* Wg       = (const float*)d_in[10];
  const float* bg       = (const float*)d_in[11];
  const float* Wo       = (const float*)d_in[12];
  const float* bo       = (const float*)d_in[13];
  const float* ln1_g    = (const float*)d_in[14];
  const float* ln1_b    = (const float*)d_in[15];
  const float* Wi       = (const float*)d_in[16];
  const float* bi       = (const float*)d_in[17];
  const float* Wf_p     = (const float*)d_in[18];
  const float* bf_p     = (const float*)d_in[19];
  const float* ln2_g    = (const float*)d_in[20];
  const float* ln2_b    = (const float*)d_in[21];
  const float* p1_w     = (const float*)d_in[22];
  const float* p1_b     = (const float*)d_in[23];
  const float* pln_g    = (const float*)d_in[24];
  const float* pln_b    = (const float*)d_in[25];
  const float* p2_w     = (const float*)d_in[26];
  const float* p2_b     = (const float*)d_in[27];

  char* ws = (char*)d_ws;
  size_t off = 0;
  auto alloc = [&](size_t bytes) -> char* {
    char* p = ws + off;
    off = (off + bytes + 255) & ~(size_t)255;
    return p;
  };
  float*    h_f32   = (float*)   alloc((size_t)kTok * kHD * 4);
  float*    pre_f32 = (float*)   alloc((size_t)kTok * kHD * 4);
  uint16_t* h_bf    = (uint16_t*)alloc((size_t)kTok * kHD * 2);
  uint16_t* qkv_bf  = (uint16_t*)alloc((size_t)kTok * kQKV * 2);
  uint16_t* gkv_bf  = (uint16_t*)alloc((size_t)kTok * kGKV * 2);
  uint16_t* attn_bf = (uint16_t*)alloc((size_t)kTok * kHD * 2);
  uint16_t* ff_bf   = (uint16_t*)alloc((size_t)kTok * kFF * 2);
  float*    qg      = (float*)   alloc((size_t)kB * kHD * 4);
  float*    biasq   = (float*)   alloc((size_t)kL * kQKV * 4);
  uint16_t* wqkvT   = (uint16_t*)alloc((size_t)kL * kQKV * kHD * 2);
  uint16_t* wgT     = (uint16_t*)alloc((size_t)kL * kGKV * kHD * 2);
  uint16_t* woT     = (uint16_t*)alloc((size_t)kL * kHD * kHD * 2);
  uint16_t* wiT     = (uint16_t*)alloc((size_t)kL * kFF * kHD * 2);
  uint16_t* wfT     = (uint16_t*)alloc((size_t)kL * kHD * kFF * 2);

  const int rb_hd = (kHD * kHD + 255) / 256;
  const int rb_ff = (kHD * kFF + 255) / 256;
  for (int l = 0; l < kL; ++l) {
    for (int i = 0; i < 3; ++i)
      repack_kernel<<<rb_hd, 256, 0, stream>>>(
          Wqkv + (size_t)(l * 3 + i) * kHD * kHD,
          wqkvT + (size_t)l * kQKV * kHD + (size_t)i * kHD * kHD, kHD, kHD,
          i == 0 ? kScale : 1.f);
    for (int i = 1; i < 3; ++i)
      repack_kernel<<<rb_hd, 256, 0, stream>>>(
          Wg + (size_t)(l * 3 + i) * kHD * kHD,
          wgT + (size_t)l * kGKV * kHD + (size_t)(i - 1) * kHD * kHD, kHD, kHD, 1.f);
    repack_kernel<<<rb_hd, 256, 0, stream>>>(
        Wo + (size_t)l * kHD * kHD, woT + (size_t)l * kHD * kHD, kHD, kHD, 1.f);
    repack_kernel<<<rb_ff, 256, 0, stream>>>(
        Wi + (size_t)l * kHD * kFF, wiT + (size_t)l * kFF * kHD, kHD, kFF, 1.f);
    repack_kernel<<<rb_ff, 256, 0, stream>>>(
        Wf_p + (size_t)l * kFF * kHD, wfT + (size_t)l * kHD * kFF, kFF, kHD, 1.f);
  }
  qkv_bias_kernel<<<(kL * kQKV + 255) / 256, 256, 0, stream>>>(bqkv, biasq);

  embed_kernel<<<kTok, 256, 0, stream>>>(x, emb_w, emb_b, pos_emb, hf_pos, tok_emb, pre_f32);
  ln_kernel<<<kTok, 256, 0, stream>>>(pre_f32, (const float*)nullptr, eln_g, eln_b,
                                      1e-12f, h_f32, h_bf);

  const dim3 gQKV(kQKV / 256, kTok / 64);
  const dim3 gGKV(kGKV / 256, kTok / 64);
  const dim3 gHD (kHD  / 256, kTok / 64);
  const dim3 gFF (kFF  / 256, kTok / 64);
  for (int l = 0; l < kL; ++l) {
    gemm_bf16<1><<<gQKV, 256, 0, stream>>>(h_bf, wqkvT + (size_t)l * kQKV * kHD,
                                           biasq + l * kQKV, qkv_bf, kTok, kQKV, kHD);
    gemm_bf16<1><<<gGKV, 256, 0, stream>>>(h_bf, wgT + (size_t)l * kGKV * kHD,
                                           bg + (size_t)l * 3 * kHD + kHD, gkv_bf,
                                           kTok, kGKV, kHD);
    qg_kernel<<<kB, kHD, 0, stream>>>(h_f32, Wg + (size_t)l * 3 * kHD * kHD,
                                      bg + (size_t)l * 3 * kHD, qg);
    attn_local<<<dim3(kS / 128, kNH, kB), 128, 0, stream>>>(qkv_bf, attn_bf);
    attn_global<<<dim3(kNH, kB), 256, 0, stream>>>(qg, gkv_bf, attn_bf);
    gemm_bf16<0><<<gHD, 256, 0, stream>>>(attn_bf, woT + (size_t)l * kHD * kHD,
                                          bo + (size_t)l * kHD, pre_f32, kTok, kHD, kHD);
    ln_kernel<<<kTok, 256, 0, stream>>>(h_f32, pre_f32, ln1_g + (size_t)l * kHD,
                                        ln1_b + (size_t)l * kHD, 1e-12f, h_f32, h_bf);
    gemm_bf16<2><<<gFF, 256, 0, stream>>>(h_bf, wiT + (size_t)l * kFF * kHD,
                                          bi + (size_t)l * kFF, ff_bf, kTok, kFF, kHD);
    gemm_bf16<0><<<gHD, 256, 0, stream>>>(ff_bf, wfT + (size_t)l * kHD * kFF,
                                          bf_p + (size_t)l * kHD, pre_f32, kTok, kHD, kFF);
    ln_kernel<<<kTok, 256, 0, stream>>>(h_f32, pre_f32, ln2_g + (size_t)l * kHD,
                                        ln2_b + (size_t)l * kHD, 1e-12f, h_f32, h_bf);
  }
  head_kernel<<<kB, 256, 0, stream>>>(h_f32, p1_w, p1_b, pln_g, pln_b, p2_w, p2_b,
                                      (float*)d_out);
}
